// ConvWithRowEncoder_2173253452015
// MI455X (gfx1250) — matrix operand order repaired and checked by reading, not compiled, not run on hardware
//
#include <hip/hip_runtime.h>
#include <math.h>

// ---------------------------------------------------------------------------
// Types for CDNA5 WMMA (wave32)
// ---------------------------------------------------------------------------
typedef __attribute__((ext_vector_type(16))) __bf16 v16bf;
typedef __attribute__((ext_vector_type(8)))  float  v8f;

union Frag16 { v16bf v; unsigned int u[8]; };

__device__ __forceinline__ unsigned short f2bf(float f) {
    unsigned int u = __float_as_uint(f);
    unsigned int r = (u + 0x7FFFu + ((u >> 16) & 1u)) >> 16;   // RNE
    return (unsigned short)r;
}
__device__ __forceinline__ float bf2f(unsigned short h) {
    return __uint_as_float(((unsigned int)h) << 16);
}
__device__ __forceinline__ float sigmoidf(float x) {
    return 1.0f / (1.0f + __expf(-x));
}

// ---------------------------------------------------------------------------
// conv1: 1x1 conv, 3 -> 64 ch, f32 NCHW input -> bf16 NHWC output
// ---------------------------------------------------------------------------
__global__ void k_conv1(const float* __restrict__ x, const float* __restrict__ w,
                        const float* __restrict__ b, unsigned short* __restrict__ out,
                        int N, int H, int W) {
    long long idx = (long long)blockIdx.x * 256 + threadIdx.x;
    long long total = (long long)N * H * W * 64;
    if (idx >= total) return;
    int oc = (int)(idx & 63);
    long long p = idx >> 6;                 // n*H*W + h*W + w
    int wq = (int)(p % W); long long t = p / W;
    int h  = (int)(t % H); int n = (int)(t / H);
    const float* xb = x + (long long)n * 3 * H * W + (long long)h * W + wq;
    long long hw = (long long)H * W;
    float acc = b[oc] + xb[0] * w[oc*3+0] + xb[hw] * w[oc*3+1] + xb[2*hw] * w[oc*3+2];
    out[idx] = f2bf(acc);
}

// ---------------------------------------------------------------------------
// Weight reorder: OIHW f32 -> [OC][KH*KW*IC] bf16 (ic fastest)
// ---------------------------------------------------------------------------
__global__ void k_prep_wt(const float* __restrict__ w, unsigned short* __restrict__ o,
                          int OC, int IC, int KH, int KW) {
    int K = KH * KW * IC;
    long long idx = (long long)blockIdx.x * 256 + threadIdx.x;
    if (idx >= (long long)OC * K) return;
    int oc = (int)(idx / K); int k = (int)(idx % K);
    int ic = k % IC; int kw = (k / IC) % KW; int kh = k / (IC * KW);
    o[idx] = f2bf(w[(((long long)oc * IC + ic) * KH + kh) * KW + kw]);
}

// BN + bias fusion: scale = g/sqrt(v+eps); shift = bias*scale + b - m*scale
__global__ void k_prep_ss(const float* __restrict__ bias, const float* __restrict__ g,
                          const float* __restrict__ bb, const float* __restrict__ m,
                          const float* __restrict__ v, float* __restrict__ scale,
                          float* __restrict__ shift, int C) {
    int c = blockIdx.x * 256 + threadIdx.x;
    if (c >= C) return;
    float s = g[c] * rsqrtf(v[c] + 1e-5f);
    scale[c] = s;
    shift[c] = bias[c] * s + bb[c] - m[c] * s;
}

// LSTM weight prep: permute gate rows (new_row 4j+g <- old g*256+j), f32->bf16
__global__ void k_prep_lstm(const float* __restrict__ wih, const float* __restrict__ whh,
                            const float* __restrict__ bih, const float* __restrict__ bhh,
                            unsigned short* __restrict__ wihp, unsigned short* __restrict__ whhp,
                            float* __restrict__ biasp) {
    int idx = blockIdx.x * 256 + threadIdx.x;
    const int tot1 = 1024 * 512, tot2 = 1024 * 256;
    if (idx < tot1) {
        int n = idx >> 9, k = idx & 511;
        int gate = n & 3, j = n >> 2;
        wihp[idx] = f2bf(wih[(gate * 256 + j) * 512 + k]);
    } else if (idx < tot1 + tot2) {
        int r = idx - tot1; int n = r >> 8, k = r & 255;
        int gate = n & 3, j = n >> 2;
        whhp[r] = f2bf(whh[(gate * 256 + j) * 256 + k]);
    } else if (idx < tot1 + tot2 + 1024) {
        int n = idx - tot1 - tot2; int gate = n & 3, j = n >> 2;
        biasp[n] = bih[gate * 256 + j] + bhh[gate * 256 + j];
    }
}

// ---------------------------------------------------------------------------
// Generic implicit-GEMM conv (VALID), bf16 NHWC in, bf16 or f32 out.
// Weights: bf16 [OC][KH*KW*IC] (ic fastest). IC must be a multiple of 32.
// Block = 256 thr (8 waves). Block tile: 64 output pixels x 128 output chans.
// Each wave: 4 M-tiles x 1 N-tile. Double-buffered LDS staging, branchless
// loads (invalid rows read a dummy in-bounds address; their D rows are
// discarded by the epilogue guard), one barrier per K-step, flattened K loop.
// ---------------------------------------------------------------------------
#define AROW 20   // padded LDS row stride in dwords (16B-aligned, conflict-reduced)

__global__ __launch_bounds__(256)
void k_wmma_conv(const unsigned short* __restrict__ in, const unsigned short* __restrict__ wt,
                 const float* __restrict__ scale, const float* __restrict__ shift,
                 unsigned short* __restrict__ obf, float* __restrict__ of32,
                 int Nimg, int IH, int IW, int IC, int OH, int OW, int OC, int KH, int KW) {
    __shared__ unsigned int Atile[2][64 * AROW];  // double-buffered A stage
    const int  M    = Nimg * OH * OW;
    const int  mt   = blockIdx.x;                 // tile of 64 pixels
    const int  lane = threadIdx.x & 31;
    const int  wave = threadIdx.x >> 5;
    const int  hi   = lane >> 4;
    const int  mrow = lane & 15;
    const int  oc   = blockIdx.y * 128 + wave * 16 + mrow;
    const int  K    = KH * KW * IC;
    const unsigned short* wrow = wt + (long long)oc * K;   // this lane's B column

    // cooperative A staging roles: 4 rows per thread, branchless addresses
    const int srow = threadIdx.x >> 4;            // 0..15
    const int kp   = threadIdx.x & 15;            // pair 0..15
    const unsigned short* pb[4];
#pragma unroll
    for (int j = 0; j < 4; ++j) {
        long long m_g = (long long)mt * 64 + j * 16 + srow;
        if (m_g < M) {
            int ow = (int)(m_g % OW); long long t2 = m_g / OW;
            int oh = (int)(t2 % OH); int nimg = (int)(t2 / OH);
            pb[j] = in + (((long long)nimg * IH + oh) * IW + ow) * IC + 2 * kp;
        } else {
            pb[j] = in + 2 * kp;                  // dummy in-bounds; row discarded later
        }
    }

    const int S   = K >> 5;                       // K-steps of 32
    const int ICq = IC >> 5;
    const long long rowskip = (long long)(IW - KW) * IC;

    // prologue: stage step 0 into buffer 0
#pragma unroll
    for (int j = 0; j < 4; ++j)
        Atile[0][(j * 16 + srow) * AROW + kp] = *(const unsigned int*)(pb[j]);
    __syncthreads();

    v8f acc[4] = {};
    long long aoff = 32;    // activation offset of the NEXT stage
    int icq = 1, kwq = 0;
    if (icq == ICq) { icq = 0; if (++kwq == KW) { kwq = 0; aoff += rowskip; } }
    int koff = 0;

    for (int s = 0; s < S; ++s) {
        const int cur = s & 1, nxt = cur ^ 1;
        const bool have_next = (s + 1 < S);

        // prefetch next stage into registers (latency hides under WMMAs)
        unsigned int nv[4];
        if (have_next) {
#pragma unroll
            for (int j = 0; j < 4; ++j)
                nv[j] = *(const unsigned int*)(pb[j] + aoff);
            aoff += 32;
            if (++icq == ICq) { icq = 0; if (++kwq == KW) { kwq = 0; aoff += rowskip; } }
        }

        // B fragment for current step
        Frag16 bf_;
#pragma unroll
        for (int v = 0; v < 8; ++v)                       // B 16-bit layout
            bf_.u[v] = *(const unsigned int*)(wrow + koff + hi * 16 + 2 * v);
        koff += 32;

        // 4 M-tiles against one B fragment
#pragma unroll
        for (int j = 0; j < 4; ++j) {
            Frag16 af;
#pragma unroll
            for (int v = 0; v < 8; ++v) {                 // A 16-bit layout
                int kidx = v + hi * 8;
                af.u[v] = Atile[cur][(j * 16 + mrow) * AROW + kidx];
            }
            acc[j] = __builtin_amdgcn_wmma_f32_16x16x32_bf16(
                false, af.v, false, bf_.v, (short)0, acc[j], false, false);
        }

        if (have_next) {
#pragma unroll
            for (int j = 0; j < 4; ++j)
                Atile[nxt][(j * 16 + srow) * AROW + kp] = nv[j];
        }
        __syncthreads();
    }

    float sc = scale ? scale[oc] : 1.0f;
    float sh = shift[oc];
#pragma unroll
    for (int j = 0; j < 4; ++j) {
#pragma unroll
        for (int r = 0; r < 8; ++r) {
            long long mg = (long long)mt * 64 + j * 16 + r + 8 * hi;   // D layout
            if (mg < M) {
                float vv = acc[j][r] * sc + sh;
                long long oidx = mg * OC + oc;
                if (of32) of32[oidx] = vv;
                else      obf[oidx]  = f2bf(vv);
            }
        }
    }
}

// ---------------------------------------------------------------------------
// Generic maxpool on bf16 NHWC
// ---------------------------------------------------------------------------
__global__ void k_maxpool(const unsigned short* __restrict__ in, unsigned short* __restrict__ out,
                          int N, int IH, int IW, int C, int OH, int OW,
                          int kh, int kw, int sh, int sw) {
    long long idx = (long long)blockIdx.x * 256 + threadIdx.x;
    long long total = (long long)N * OH * OW * C;
    if (idx >= total) return;
    int c = (int)(idx % C); long long t = idx / C;
    int ow = (int)(t % OW); t /= OW;
    int oh = (int)(t % OH); int n = (int)(t / OH);
    float m = -3.0e38f;
    for (int i = 0; i < kh; ++i)
        for (int j = 0; j < kw; ++j) {
            unsigned short hv = in[(((long long)n * IH + (oh * sh + i)) * IW + (ow * sw + j)) * C + c];
            m = fmaxf(m, bf2f(hv));
        }
    out[idx] = f2bf(m);
}

// ---------------------------------------------------------------------------
// Bidirectional LSTM recurrence. Grid (R/16, 2 dirs), block 256 (8 waves).
// pre: f32 [R][T][1024] gate-interleaved (unit j -> cols 4j+{i,f,g,o}).
// Uw : bf16 [1024][256] row-permuted likewise. h in LDS (bf16), c in VGPRs.
// ---------------------------------------------------------------------------
__global__ __launch_bounds__(256)
void k_lstm(const float* __restrict__ preF, const float* __restrict__ preB,
            const unsigned short* __restrict__ UwF, const unsigned short* __restrict__ UwB,
            float* __restrict__ out) {
    const int T = 122;
    const int dir = blockIdx.y;
    const float* pre = dir ? preB : preF;
    const unsigned short* Uw = dir ? UwB : UwF;
    const int lane = threadIdx.x & 31, wave = threadIdx.x >> 5, hi = lane >> 4;
    const int rbase = blockIdx.x * 16;

    __shared__ unsigned short hbuf[16 * 256];     // h, bf16
    for (int i = threadIdx.x; i < 16 * 128; i += 256) ((unsigned int*)hbuf)[i] = 0u;

    float cst[8][8];
#pragma unroll
    for (int a = 0; a < 8; ++a)
#pragma unroll
        for (int b = 0; b < 8; ++b) cst[a][b] = 0.0f;
    __syncthreads();

    for (int s = 0; s < T; ++s) {
        int t = dir ? (T - 1 - s) : s;
        v8f acc[8];
#pragma unroll
        for (int nt = 0; nt < 8; ++nt) {
            int col = wave * 128 + nt * 16 + (lane & 15);
            v8f a = {};
#pragma unroll
            for (int r = 0; r < 8; ++r) {
                int row = r + 8 * hi;
                a[r] = pre[((long long)(rbase + row) * T + t) * 1024 + col];
            }
#pragma unroll
            for (int ks = 0; ks < 8; ++ks) {
                Frag16 af, bf_;
                int mrow = lane & 15;
#pragma unroll
                for (int v = 0; v < 8; ++v) {
                    int kloc = 2 * v + hi * 16 + ks * 32;
                    af.u[v] = *(const unsigned int*)(hbuf + mrow * 256 + kloc);
                    int kg = ks * 32 + hi * 16 + 2 * v;
                    bf_.u[v] = *(const unsigned int*)(Uw + (long long)col * 256 + kg);
                }
                a = __builtin_amdgcn_wmma_f32_16x16x32_bf16(false, af.v, false, bf_.v,
                                                            (short)0, a, false, false);
            }
            acc[nt] = a;
        }
        __syncthreads();   // everyone done reading h

#pragma unroll
        for (int nt = 0; nt < 8; ++nt) {
            int col  = wave * 128 + nt * 16 + (lane & 15);
            int unit = col >> 2;
            int q    = lane & 3;
#pragma unroll
            for (int r = 0; r < 8; ++r) {
                float xv = acc[nt][r];
                float x1 = __shfl_xor(xv, 1);
                float x2 = __shfl_xor(xv, 2);
                float x3 = __shfl_xor(xv, 3);
                float gg[4];
                gg[q] = xv; gg[q ^ 1] = x1; gg[q ^ 2] = x2; gg[q ^ 3] = x3;
                float gi = gg[0], gf = gg[1], gc = gg[2], go = gg[3];
                float c = sigmoidf(gf) * cst[nt][r] + sigmoidf(gi) * tanhf(gc);
                cst[nt][r] = c;
                float h = sigmoidf(go) * tanhf(c);
                int row = r + 8 * hi;
                if (q == 0) {
                    hbuf[row * 256 + unit] = f2bf(h);
                    int grow = rbase + row;
                    int n = grow / 10, wi = grow % 10;
                    out[((long long)n * 1220 + wi * 122 + t) * 512 + dir * 256 + unit] = h;
                }
            }
        }
        __syncthreads();   // new h visible before next step
    }
}

// ---------------------------------------------------------------------------
// Host launcher
// ---------------------------------------------------------------------------
extern "C" void kernel_launch(void* const* d_in, const int* in_sizes, int n_in,
                              void* d_out, int out_size, void* d_ws, size_t ws_size,
                              hipStream_t stream) {
    (void)in_sizes; (void)n_in; (void)out_size; (void)ws_size;
    const float* x    = (const float*)d_in[0];
    const float* c1w  = (const float*)d_in[1];  const float* c1b = (const float*)d_in[2];
    const float* c2w  = (const float*)d_in[3];  const float* c2b = (const float*)d_in[4];
    const float* c3w  = (const float*)d_in[5];  const float* c3b = (const float*)d_in[6];
    const float* bn1g = (const float*)d_in[7];  const float* bn1b = (const float*)d_in[8];
    const float* bn1m = (const float*)d_in[9];  const float* bn1v = (const float*)d_in[10];
    const float* c4w  = (const float*)d_in[11]; const float* c4b = (const float*)d_in[12];
    const float* c5w  = (const float*)d_in[13]; const float* c5b = (const float*)d_in[14];
    const float* bn2g = (const float*)d_in[15]; const float* bn2b = (const float*)d_in[16];
    const float* bn2m = (const float*)d_in[17]; const float* bn2v = (const float*)d_in[18];
    const float* c6w  = (const float*)d_in[19]; const float* c6b = (const float*)d_in[20];
    const float* bn3g = (const float*)d_in[21]; const float* bn3b = (const float*)d_in[22];
    const float* bn3m = (const float*)d_in[23]; const float* bn3v = (const float*)d_in[24];
    const float* wihF = (const float*)d_in[25]; const float* whhF = (const float*)d_in[26];
    const float* bihF = (const float*)d_in[27]; const float* bhhF = (const float*)d_in[28];
    const float* wihB = (const float*)d_in[29]; const float* whhB = (const float*)d_in[30];
    const float* bihB = (const float*)d_in[31]; const float* bhhB = (const float*)d_in[32];
    float* outp = (float*)d_out;

    char* ws = (char*)d_ws;
    const size_t SLAB_A = 0;
    const size_t SZ_A   = 134217728ull;           // 128 MB
    const size_t SLAB_B = SZ_A;
    const size_t SZ_B   = 142606336ull;           // 136 MB
    size_t cur = SZ_A + SZ_B;
    auto bump = [&](size_t bytes) { size_t o = cur; cur += (bytes + 255) & ~(size_t)255; return o; };

    size_t w2o = bump(73728ull * 2),  w3o = bump(294912ull * 2), w4o = bump(589824ull * 2);
    size_t w5o = bump(1179648ull * 2), w6o = bump(2359296ull * 2);
    size_t sc3 = bump(256 * 4), sh3 = bump(256 * 4);
    size_t sc5 = bump(512 * 4), sh5 = bump(512 * 4);
    size_t sc6 = bump(512 * 4), sh6 = bump(512 * 4);
    size_t wipF = bump(524288ull * 2), whpF = bump(262144ull * 2), bpF = bump(1024 * 4);
    size_t wipB = bump(524288ull * 2), whpB = bump(262144ull * 2), bpB = bump(1024 * 4);

    auto US = [&](size_t o) { return (unsigned short*)(ws + o); };
    auto FP = [&](size_t o) { return (float*)(ws + o); };

    unsigned short* a0 = US(SLAB_A);
    unsigned short* a1 = US(SLAB_B);
    unsigned short* a2 = US(SLAB_A);
    unsigned short* a3 = US(SLAB_B);
    unsigned short* a4 = US(SLAB_A);
    unsigned short* a5 = US(SLAB_B);
    unsigned short* a6 = US(SLAB_A);
    unsigned short* a7 = US(SLAB_B);
    unsigned short* a8 = US(SLAB_A);
    float* preF = FP(SLAB_B);
    float* preB = FP(SLAB_A + 33554432ull);       // after a8 (20 MB)

    // ---- weight / epilogue prep ----
    k_prep_wt<<<(128 * 576  + 255) / 256, 256, 0, stream>>>(c2w, US(w2o), 128, 64, 3, 3);
    k_prep_wt<<<(256 * 1152 + 255) / 256, 256, 0, stream>>>(c3w, US(w3o), 256, 128, 3, 3);
    k_prep_wt<<<(256 * 2304 + 255) / 256, 256, 0, stream>>>(c4w, US(w4o), 256, 256, 3, 3);
    k_prep_wt<<<(512 * 2304 + 255) / 256, 256, 0, stream>>>(c5w, US(w5o), 512, 256, 3, 3);
    k_prep_wt<<<(512 * 4608 + 255) / 256, 256, 0, stream>>>(c6w, US(w6o), 512, 512, 3, 3);
    k_prep_ss<<<1, 256, 0, stream>>>(c3b, bn1g, bn1b, bn1m, bn1v, FP(sc3), FP(sh3), 256);
    k_prep_ss<<<2, 256, 0, stream>>>(c5b, bn2g, bn2b, bn2m, bn2v, FP(sc5), FP(sh5), 512);
    k_prep_ss<<<2, 256, 0, stream>>>(c6b, bn3g, bn3b, bn3m, bn3v, FP(sc6), FP(sh6), 512);
    int lstm_tot = 1024 * 512 + 1024 * 256 + 1024;
    k_prep_lstm<<<(lstm_tot + 255) / 256, 256, 0, stream>>>(wihF, whhF, bihF, bhhF, US(wipF), US(whpF), FP(bpF));
    k_prep_lstm<<<(lstm_tot + 255) / 256, 256, 0, stream>>>(wihB, whhB, bihB, bhhB, US(wipB), US(whpB), FP(bpB));

    // ---- conv chain ----
    k_conv1<<<(16ll * 64 * 512 * 64 + 255) / 256, 256, 0, stream>>>(x, c1w, c1b, a0, 16, 64, 512);

    // conv2: (16,64,512,64) -> (16,62,510,128)
    k_wmma_conv<<<dim3((16 * 62 * 510 + 63) / 64, 1), 256, 0, stream>>>(
        a0, US(w2o), nullptr, c2b, a1, nullptr, 16, 64, 512, 64, 62, 510, 128, 3, 3);
    // pool 2x2 s2 -> (16,31,255,128)
    k_maxpool<<<(16ll * 31 * 255 * 128 + 255) / 256, 256, 0, stream>>>(
        a1, a2, 16, 62, 510, 128, 31, 255, 2, 2, 2, 2);
    // conv3 + bn1: -> (16,29,253,256)
    k_wmma_conv<<<dim3((16 * 29 * 253 + 63) / 64, 2), 256, 0, stream>>>(
        a2, US(w3o), FP(sc3), FP(sh3), a3, nullptr, 16, 31, 255, 128, 29, 253, 256, 3, 3);
    // conv4: -> (16,27,251,256)
    k_wmma_conv<<<dim3((16 * 27 * 251 + 63) / 64, 2), 256, 0, stream>>>(
        a3, US(w4o), nullptr, c4b, a4, nullptr, 16, 29, 253, 256, 27, 251, 256, 3, 3);
    // pool 2x2 s1 -> (16,26,250,256)
    k_maxpool<<<(16ll * 26 * 250 * 256 + 255) / 256, 256, 0, stream>>>(
        a4, a5, 16, 27, 251, 256, 26, 250, 2, 2, 1, 1);
    // conv5 + bn2: -> (16,24,248,512)
    k_wmma_conv<<<dim3((16 * 24 * 248 + 63) / 64, 4), 256, 0, stream>>>(
        a5, US(w5o), FP(sc5), FP(sh5), a6, nullptr, 16, 26, 250, 256, 24, 248, 512, 3, 3);
    // pool 1x1 s2 -> (16,12,124,512)
    k_maxpool<<<(16ll * 12 * 124 * 512 + 255) / 256, 256, 0, stream>>>(
        a6, a7, 16, 24, 248, 512, 12, 124, 1, 1, 2, 2);
    // conv6 + bn3: -> (16,10,122,512) == rows (160,122,512)
    k_wmma_conv<<<dim3((16 * 10 * 122 + 63) / 64, 4), 256, 0, stream>>>(
        a7, US(w6o), FP(sc6), FP(sh6), a8, nullptr, 16, 12, 124, 512, 10, 122, 512, 3, 3);

    // ---- LSTM input projections (1x1 "conv": M=160*122, K=512, N=1024, f32 out) ----
    k_wmma_conv<<<dim3((160 * 122 + 63) / 64, 8), 256, 0, stream>>>(
        a8, US(wipF), nullptr, FP(bpF), nullptr, preF, 160, 122, 1, 512, 122, 1, 1024, 1, 1);
    k_wmma_conv<<<dim3((160 * 122 + 63) / 64, 8), 256, 0, stream>>>(
        a8, US(wipB), nullptr, FP(bpB), nullptr, preB, 160, 122, 1, 512, 122, 1, 1024, 1, 1);

    // ---- LSTM recurrence, both directions ----
    k_lstm<<<dim3(10, 2), 256, 0, stream>>>(preF, preB, US(whpF), US(whpB), outp);
}
